// NSR_11166914969869
// MI455X (gfx1250) — matrix operand order repaired and checked by reading, not compiled, not run on hardware
//
#include <hip/hip_runtime.h>
#include <hip/hip_bf16.h>

// ---------------------------------------------------------------------------
// NSR oscillatory SSM forward for MI455X (gfx1250).
// Layer-pipelined persistent kernel: 4 workgroups (one per layer, each 32
// wave32s on its own WGP). u activations (16x256 f32 = 16KB) flow through an
// L2 ring buffer with acquire/release flags; weights are pre-packed to f16
// WMMA B-fragment layout and streamed from L2 each step. All GEMMs use
// v_wmma_f32_16x16x32_f16 with M=16 (batch) matching the WMMA tile.
// k-loops are unrolled only 2x to keep the live B-fragment set inside the
// per-wave VGPR budget (8 waves/SIMD) -> no scratch spills. Activations use
// v_rcp_f32/v_rsq_f32 fast paths instead of IEEE div expansions.
// ---------------------------------------------------------------------------

#define B_ 16
#define C_ 256
#define T_ 2048
#define H_ 512
#define NPAIR_ 256
#define DEPTH 16           // ring-buffer depth per pipeline boundary
#define EPS_ 1e-5f

typedef _Float16 v16h __attribute__((ext_vector_type(16)));
typedef float    v8f  __attribute__((ext_vector_type(8)));

__device__ __forceinline__ v8f wmma16(v16h a, v16h b, v8f c) {
  return __builtin_amdgcn_wmma_f32_16x16x32_f16(
      /*neg_a=*/false, a, /*neg_b=*/false, b,
      /*c_mod=*/(short)0, c, /*reuse_a=*/false, /*reuse_b=*/false);
}

// silu via v_exp + v_rcp (1 ulp), no IEEE div expansion
__device__ __forceinline__ float silu_f(float x) {
  return x * __builtin_amdgcn_rcpf(1.f + __expf(-x));
}
// tanh(x) = 1 - 2/(exp(2x)+1), via v_exp + v_rcp
__device__ __forceinline__ float tanh_f(float x) {
  return 1.f - 2.f * __builtin_amdgcn_rcpf(__expf(2.f * x) + 1.f);
}

// Pack a (16 x ld) f32 LDS matrix into WMMA A-fragment layout (ISA 7.12.2,
// 16-bit A 16x32): lane<16: elems 0..7 -> K 0..7, 8..15 -> K 16..23;
// lane>=16: elems 0..7 -> K 8..15, 8..15 -> K 24..31. Each unit copies 8
// consecutive K values, so both LDS read and write vectorize.
__device__ __forceinline__ void packA_f32(const float* __restrict__ src, int ld,
                                          _Float16* __restrict__ dst, int KT, int tid) {
  int total = KT * 64;  // 32 lanes * 2 halves per k-tile
  for (int u2 = tid; u2 < total; u2 += 1024) {
    int half = u2 & 1, ln = (u2 >> 1) & 31, kt = u2 >> 6;
    int mm = ln & 15;
    int k0 = kt * 32 + ((ln >> 4) << 3) + (half << 4);
    _Float16* d = dst + (kt * 32 + ln) * 16 + half * 8;
    const float* sp = src + mm * ld + k0;
#pragma unroll
    for (int i = 0; i < 8; i++) d[i] = (_Float16)sp[i];
  }
}

__device__ __forceinline__ void packA_f16(const _Float16* __restrict__ src, int ld,
                                          _Float16* __restrict__ dst, int KT, int tid) {
  int total = KT * 64;
  for (int u2 = tid; u2 < total; u2 += 1024) {
    int half = u2 & 1, ln = (u2 >> 1) & 31, kt = u2 >> 6;
    int mm = ln & 15;
    int k0 = kt * 32 + ((ln >> 4) << 3) + (half << 4);
    _Float16* d = dst + (kt * 32 + ln) * 16 + half * 8;
    const _Float16* sp = src + mm * ld + k0;
#pragma unroll
    for (int i = 0; i < 8; i++) d[i] = sp[i];
  }
}

// ---------------------------------------------------------------------------
// Prep: pack a (K x N)-effective f32 weight into f16 WMMA B-fragment tiles.
// B 32x16 layout: j<8 -> K = 8*(lane>>4)+j, j>=8 -> K = 16+8*(lane>>4)+(j-8), N = lane&15.
// Packed index = (((l*NT+nt)*KT+kt)*32+lane)*16 + j  (each lane's 16 f16 are
// contiguous -> 32B per lane -> coalesced b128 pairs at load time).
// ---------------------------------------------------------------------------
__global__ void nsr_pack_b(const float* __restrict__ src, _Float16* __restrict__ dst,
                           int NT, int KT, int N, int K, int transposed, int total) {
  int id = blockIdx.x * blockDim.x + threadIdx.x;
  if (id >= total) return;
  int j = id & 15;
  int lane = (id >> 4) & 31;
  int rest = id >> 9;
  int kt = rest % KT; rest /= KT;
  int nt = rest % NT;
  int l  = rest / NT;
  int k = kt * 32 + ((lane >> 4) << 3) + (j & 7) + ((j >> 3) << 4);
  int n = nt * 16 + (lane & 15);
  float v = transposed ? src[((size_t)l * N + n) * K + k]
                       : src[((size_t)l * K + k) * N + n];
  dst[id] = (_Float16)v;
}

__global__ void nsr_prep_misc(const float* __restrict__ log_alpha,
                              const float* __restrict__ omega,
                              float* __restrict__ dcs, unsigned int* __restrict__ syncw) {
  int i = blockIdx.x * blockDim.x + threadIdx.x;
  if (i < 32) syncw[i] = 0u;            // reset pipeline flags every launch
  if (i < 4 * NPAIR_) {
    int l = i >> 8, p = i & 255;
    float la = log_alpha[i];
    float alpha = (la > 20.f) ? la : log1pf(__expf(la));  // softplus
    float th = omega[i];                                   // DT = 1
    dcs[(l * 3 + 0) * NPAIR_ + p] = __expf(-alpha);        // decay
    dcs[(l * 3 + 1) * NPAIR_ + p] = cosf(th);
    dcs[(l * 3 + 2) * NPAIR_ + p] = sinf(th);
  }
}

// ---------------------------------------------------------------------------
// Main pipeline kernel: blockIdx.x = stage = layer (0..3). Stage 3 also does
// the rW readout. 1024 threads = 32 wave32s; wave w owns output columns
// [16w, 16w+16) of each GEMM.
// ---------------------------------------------------------------------------
__global__ void __launch_bounds__(1024, 1)
nsr_stage_kernel(const float* __restrict__ x,
                 const _Float16* __restrict__ w1p, const _Float16* __restrict__ bp,
                 const _Float16* __restrict__ w2p, const _Float16* __restrict__ owp,
                 const _Float16* __restrict__ rwp,
                 const float* __restrict__ gb1, const float* __restrict__ gb2,
                 const float* __restrict__ biasL, const float* __restrict__ hsc,
                 const float* __restrict__ ob, const float* __restrict__ rsc,
                 const float* __restrict__ rb, const float* __restrict__ dcs,
                 float* __restrict__ ring, unsigned int* __restrict__ syncw,
                 float* __restrict__ out) {
  const int stage = blockIdx.x;
  const int tid = threadIdx.x;
  const int lane = tid & 31;
  const int wv = tid >> 5;

  __shared__ float s_u[B_ * C_];                        // fp32 u (residual master)
  __shared__ alignas(32) _Float16 s_uA[8 * 32 * 16];    // u packed A-frags (K=256)
  __shared__ alignas(32) _Float16 s_t1p[B_ * H_];       // silu(u@W1+b1), plain f16
  __shared__ alignas(32) _Float16 s_tA[16 * 32 * 16];   // A-frags, K=512 (t1 then xs)
  __shared__ float s_bu[B_ * H_];
  __shared__ float s_g[B_ * H_];
  __shared__ float s_h[B_ * H_];                        // oscillator state
  __shared__ float s_y[B_ * C_];
  __shared__ float s_red[1024];
  __shared__ float s_rstd[16];
  __shared__ float s_gb1[H_], s_gb2[H_], s_bias[H_], s_hsc[H_];
  __shared__ float s_ob[C_], s_rsc[C_], s_rb[C_];
  __shared__ float s_d[NPAIR_], s_c[NPAIR_], s_sn[NPAIR_];

  for (int i = tid; i < H_; i += 1024) {
    s_gb1[i] = gb1[stage * H_ + i];  s_gb2[i] = gb2[stage * H_ + i];
    s_bias[i] = biasL[stage * H_ + i];  s_hsc[i] = hsc[stage * H_ + i];
  }
  for (int i = tid; i < C_; i += 1024) {
    s_ob[i] = ob[stage * C_ + i];  s_rsc[i] = rsc[stage * C_ + i];  s_rb[i] = rb[i];
  }
  for (int i = tid; i < NPAIR_; i += 1024) {
    s_d[i]  = dcs[stage * 768 + i];
    s_c[i]  = dcs[stage * 768 + 256 + i];
    s_sn[i] = dcs[stage * 768 + 512 + i];
  }
  for (int i = tid; i < B_ * H_; i += 1024) s_h[i] = 0.f;   // h0 = zeros
  __syncthreads();

  const int m = tid >> 6;       // batch row (0..15)
  const int q = tid & 63;       // 64 threads per row for reductions

  for (int t = 0; t < T_; ++t) {
    float* ring_in  = ring + ((size_t)(stage - 1) * DEPTH + (t & (DEPTH - 1))) * (B_ * C_);
    float* ring_out = ring + ((size_t)stage * DEPTH + (t & (DEPTH - 1))) * (B_ * C_);

    // ---- acquire u for this step -------------------------------------
    if (stage == 0) {
      for (int i = tid; i < B_ * C_; i += 1024)
        s_u[i] = x[(size_t)i * T_ + t];                  // u[b][c] = x[b][c][t]
      __syncthreads();
    } else {
      if (tid == 0) {
        while (__hip_atomic_load(&syncw[stage - 1], __ATOMIC_ACQUIRE,
                                 __HIP_MEMORY_SCOPE_AGENT) < (unsigned)(t + 1))
          __builtin_amdgcn_s_sleep(1);
      }
      __syncthreads();
      for (int i = tid; i < B_ * C_; i += 1024)
        s_u[i] = __hip_atomic_load(&ring_in[i], __ATOMIC_RELAXED, __HIP_MEMORY_SCOPE_AGENT);
      __syncthreads();
      if (tid == 0)  // publish consumption for producer back-pressure
        __hip_atomic_store(&syncw[8 + stage - 1], (unsigned)(t + 1),
                           __ATOMIC_RELAXED, __HIP_MEMORY_SCOPE_AGENT);
    }

    packA_f32(s_u, C_, s_uA, 8, tid);
    __syncthreads();

    // ---- GEMM1 (t1 = silu(u@gW1^T+gb1)) + GEMM2 (Bu = u@Bmat), N=512 --
    {
      v8f a1 = {}, a2 = {};
      const _Float16* w1t = w1p + ((size_t)(stage * 32 + wv) * 8) * 512;
      const _Float16* bt  = bp  + ((size_t)(stage * 32 + wv) * 8) * 512;
#pragma unroll 2
      for (int kt = 0; kt < 8; kt++) {
        v16h av = *(const v16h*)&s_uA[(kt * 32 + lane) * 16];
        v16h b1 = *(const v16h*)&w1t[kt * 512 + lane * 16];
        v16h b2 = *(const v16h*)&bt[kt * 512 + lane * 16];
        a1 = wmma16(av, b1, a1);
        a2 = wmma16(av, b2, a2);
      }
      int nn = wv * 16 + (lane & 15);
      int rowbase = (lane >> 4) * 8;
#pragma unroll
      for (int r = 0; r < 8; r++) {
        int row = rowbase + r;
        s_t1p[row * H_ + nn] = (_Float16)silu_f(a1[r] + s_gb1[nn]);
        s_bu[row * H_ + nn]  = a2[r];
      }
    }
    __syncthreads();

    packA_f16(s_t1p, H_, s_tA, 16, tid);
    __syncthreads();

    // ---- GEMM3: g = tanh(t1 @ gW2^T + gb2), K=512, N=512 --------------
    {
      v8f a = {};
      const _Float16* w2t = w2p + ((size_t)(stage * 32 + wv) * 16) * 512;
#pragma unroll 2
      for (int kt = 0; kt < 16; kt++) {
        v16h av = *(const v16h*)&s_tA[(kt * 32 + lane) * 16];
        v16h bv = *(const v16h*)&w2t[kt * 512 + lane * 16];
        a = wmma16(av, bv, a);
      }
      int nn = wv * 16 + (lane & 15);
      int rowbase = (lane >> 4) * 8;
#pragma unroll
      for (int r = 0; r < 8; r++)
        s_g[(rowbase + r) * H_ + nn] = tanh_f(a[r] + s_gb2[nn]);
    }
    __syncthreads();

    // ---- 2 micro steps: rotate/decay + (1+g)*Bu + bias, silu, RMS(H) --
    for (int ms = 0; ms < 2; ms++) {
      float vv[8]; float ss = 0.f;
#pragma unroll
      for (int i = 0; i < 4; i++) {
        int p = q * 4 + i;
        float d = s_d[p], cc = s_c[p], sv = s_sn[p];
        float x1 = s_h[m * H_ + 2 * p], x2 = s_h[m * H_ + 2 * p + 1];
        float xr1 = d * (cc * x1 - sv * x2);
        float xr2 = d * (sv * x1 + cc * x2);
        int h0 = 2 * p;
        float xn1 = xr1 + (1.f + s_g[m * H_ + h0])     * s_bu[m * H_ + h0]     + s_bias[h0];
        float xn2 = xr2 + (1.f + s_g[m * H_ + h0 + 1]) * s_bu[m * H_ + h0 + 1] + s_bias[h0 + 1];
        float v1 = silu_f(xn1), v2 = silu_f(xn2);
        vv[2 * i] = v1; vv[2 * i + 1] = v2;
        ss += v1 * v1 + v2 * v2;
      }
      s_red[m * 64 + q] = ss;
      __syncthreads();
      if (tid < 16) {
        float acc = 0.f;
        for (int j2 = 0; j2 < 64; j2++) acc += s_red[tid * 64 + j2];
        s_rstd[tid] = __builtin_amdgcn_rsqf(acc * (1.f / H_) + EPS_);
      }
      __syncthreads();
      float rs = s_rstd[m];
#pragma unroll
      for (int i = 0; i < 4; i++) {
        int h0 = 2 * (q * 4 + i);
        s_h[m * H_ + h0]     = vv[2 * i]     * rs * s_hsc[h0];
        s_h[m * H_ + h0 + 1] = vv[2 * i + 1] * rs * s_hsc[h0 + 1];
      }
      __syncthreads();
    }

    packA_f32(s_h, H_, s_tA, 16, tid);   // xs as A-frags for readout GEMM
    __syncthreads();

    // ---- GEMM4: y = xs @ oW^T + ob, K=512, N=256 (waves 0..15) --------
    if (wv < 16) {
      v8f a = {};
      const _Float16* owt = owp + ((size_t)(stage * 16 + wv) * 16) * 512;
#pragma unroll 2
      for (int kt = 0; kt < 16; kt++) {
        v16h av = *(const v16h*)&s_tA[(kt * 32 + lane) * 16];
        v16h bv = *(const v16h*)&owt[kt * 512 + lane * 16];
        a = wmma16(av, bv, a);
      }
      int nn = wv * 16 + (lane & 15);
      int rowbase = (lane >> 4) * 8;
#pragma unroll
      for (int r = 0; r < 8; r++)
        s_y[(rowbase + r) * C_ + nn] = a[r] + s_ob[nn];
    }
    __syncthreads();

    // ---- u = rms(u + y) over C; hand off / readout --------------------
    {
      float un[4]; float ss = 0.f;
#pragma unroll
      for (int i = 0; i < 4; i++) {
        int cc = q * 4 + i;
        un[i] = s_u[m * C_ + cc] + s_y[m * C_ + cc];
        ss += un[i] * un[i];
      }
      s_red[m * 64 + q] = ss;
      __syncthreads();
      if (tid < 16) {
        float acc = 0.f;
        for (int j2 = 0; j2 < 64; j2++) acc += s_red[tid * 64 + j2];
        s_rstd[tid] = __builtin_amdgcn_rsqf(acc * (1.f / C_) + EPS_);
      }
      __syncthreads();
      float rs = s_rstd[m];

      if (stage < 3) {
        if (tid == 0 && t >= DEPTH) {  // ring back-pressure
          while (__hip_atomic_load(&syncw[8 + stage], __ATOMIC_RELAXED,
                                   __HIP_MEMORY_SCOPE_AGENT) <= (unsigned)(t - DEPTH))
            __builtin_amdgcn_s_sleep(1);
        }
        __syncthreads();
#pragma unroll
        for (int i = 0; i < 4; i++) {
          int cc = q * 4 + i;
          ring_out[m * C_ + cc] = un[i] * rs * s_rsc[cc];
        }
        __threadfence();
        __syncthreads();
        if (tid == 0)
          __hip_atomic_store(&syncw[stage], (unsigned)(t + 1),
                             __ATOMIC_RELEASE, __HIP_MEMORY_SCOPE_AGENT);
      } else {
        // stage 3: final u -> out[:, :, t] = u @ rW^T + rb
#pragma unroll
        for (int i = 0; i < 4; i++) {
          int cc = q * 4 + i;
          s_u[m * C_ + cc] = un[i] * rs * s_rsc[cc];
        }
        __syncthreads();
        packA_f32(s_u, C_, s_uA, 8, tid);
        __syncthreads();
        if (wv < 16) {
          v8f a = {};
          const _Float16* rwt = rwp + ((size_t)wv * 8) * 512;
#pragma unroll 2
          for (int kt = 0; kt < 8; kt++) {
            v16h av = *(const v16h*)&s_uA[(kt * 32 + lane) * 16];
            v16h bv = *(const v16h*)&rwt[kt * 512 + lane * 16];
            a = wmma16(av, bv, a);
          }
          int nn = wv * 16 + (lane & 15);
          int rowbase = (lane >> 4) * 8;
#pragma unroll
          for (int r = 0; r < 8; r++)
            out[((size_t)((rowbase + r) * C_ + nn)) * T_ + t] = a[r] + s_rb[nn];
        }
      }
    }
  }
}

// ---------------------------------------------------------------------------
extern "C" void kernel_launch(void* const* d_in, const int* in_sizes, int n_in,
                              void* d_out, int out_size, void* d_ws, size_t ws_size,
                              hipStream_t stream) {
  (void)in_sizes; (void)n_in; (void)out_size; (void)ws_size;
  const float* x         = (const float*)d_in[0];
  const float* log_alpha = (const float*)d_in[1];
  const float* omega     = (const float*)d_in[2];
  const float* Bmat      = (const float*)d_in[3];
  const float* bias      = (const float*)d_in[4];
  const float* gW1       = (const float*)d_in[5];
  const float* gb1       = (const float*)d_in[6];
  const float* gW2       = (const float*)d_in[7];
  const float* gb2       = (const float*)d_in[8];
  const float* hsc       = (const float*)d_in[9];
  const float* oW        = (const float*)d_in[10];
  const float* ob        = (const float*)d_in[11];
  const float* rsc       = (const float*)d_in[12];
  const float* rW        = (const float*)d_in[13];
  const float* rb        = (const float*)d_in[14];
  float* out = (float*)d_out;

  char* ws = (char*)d_ws;
  size_t off = 0;
  auto wsalloc = [&](size_t bytes) -> void* {
    void* p = ws + off;
    off = (off + bytes + 255) & ~(size_t)255;
    return p;
  };
  unsigned int* syncw = (unsigned int*)wsalloc(256);
  float*     dcs = (float*)wsalloc((size_t)4 * 3 * 256 * 4);
  _Float16*  w1p = (_Float16*)wsalloc((size_t)524288 * 2);   // gW1^T packed
  _Float16*  bpk = (_Float16*)wsalloc((size_t)524288 * 2);   // Bmat packed
  _Float16*  w2p = (_Float16*)wsalloc((size_t)1048576 * 2);  // gW2^T packed
  _Float16*  owp = (_Float16*)wsalloc((size_t)524288 * 2);   // oW^T packed
  _Float16*  rwp = (_Float16*)wsalloc((size_t)65536 * 2);    // rW^T packed
  float*    ring = (float*)wsalloc((size_t)3 * DEPTH * (B_ * C_) * 4);

  // prep (reruns every launch: deterministic, resets pipeline flags)
  nsr_prep_misc<<<4, 256, 0, stream>>>(log_alpha, omega, dcs, syncw);
  nsr_pack_b<<<524288 / 256,  256, 0, stream>>>(gW1,  w1p, 32,  8, 512, 256, 1, 524288);
  nsr_pack_b<<<524288 / 256,  256, 0, stream>>>(Bmat, bpk, 32,  8, 512, 256, 0, 524288);
  nsr_pack_b<<<1048576 / 256, 256, 0, stream>>>(gW2,  w2p, 32, 16, 512, 512, 1, 1048576);
  nsr_pack_b<<<524288 / 256,  256, 0, stream>>>(oW,   owp, 16, 16, 256, 512, 1, 524288);
  nsr_pack_b<<<65536 / 256,   256, 0, stream>>>(rW,   rwp, 16,  8, 256, 256, 1, 65536);

  // persistent 4-stage layer pipeline (one block per WGP)
  nsr_stage_kernel<<<4, 1024, 0, stream>>>(x, w1p, bpk, w2p, owp, rwp,
                                           gb1, gb2, bias, hsc, ob, rsc, rb,
                                           dcs, ring, syncw, out);
}
